// VectorQuantizer_22488448761986
// MI455X (gfx1250) — compile-verified, matrix operand order checked by reading, not run on hardware
//
#include <hip/hip_runtime.h>
#include <hip/hip_bf16.h>

// ---------------------------------------------------------------------------
// VQ (cosine/argmax form) for MI455X gfx1250:
//   bf16 WMMA 16x16x32 main loop, TDM (tensor_load_to_lds) double-buffered
//   codebook staging, b128 LDS fragment loads.
// ---------------------------------------------------------------------------

typedef __attribute__((ext_vector_type(16))) __bf16       v16bf;
typedef __attribute__((ext_vector_type(8)))  float        v8f;
typedef __attribute__((ext_vector_type(8)))  unsigned int v8u;
typedef __attribute__((ext_vector_type(4)))  unsigned int v4ui;
typedef __attribute__((ext_vector_type(8)))  int          v8i;
typedef __attribute__((ext_vector_type(4)))  int          v4i;

union FragU { v4ui q[2]; v8u u; v16bf h; };

#define DVEC       64          // feature dim
#define DW         32          // dwords per row (64 bf16)
#define NCODES     8192
#define NTOK       32768
#define CHUNK      128         // codebook rows staged in LDS per TDM transfer
#define LDS_STRIDE 36          // padded dword stride: 144B, 16B aligned
#define CH_DWORDS  (CHUNK * LDS_STRIDE)
#define NCHUNK     (NCODES / CHUNK)

static __device__ inline unsigned short f2bf(float f) {
  unsigned int u = __float_as_uint(f);
  unsigned int r = (u + 0x7FFFu + ((u >> 16) & 1u)) >> 16;   // RNE
  return (unsigned short)r;
}

// One wave per row: L2-normalize a [rows x 64] f32 matrix into packed bf16.
__global__ __launch_bounds__(256) void nrm_rows(const float* __restrict__ src,
                                                unsigned int* __restrict__ dst,
                                                int rows, float* lossPtr) {
  const int wave = threadIdx.x >> 5;
  const int lane = threadIdx.x & 31;
  const int row  = blockIdx.x * 8 + wave;
  if (lossPtr && blockIdx.x == 0 && threadIdx.x == 0) *lossPtr = 0.0f;
  if (row >= rows) return;
  const float2 v = ((const float2*)(src + (size_t)row * DVEC))[lane];
  float ss = v.x * v.x + v.y * v.y;
#pragma unroll
  for (int m = 16; m >= 1; m >>= 1) ss += __shfl_xor(ss, m, 32);
  const float s = 1.0f / fmaxf(sqrtf(ss), 1e-12f);
  unsigned int p = (unsigned int)f2bf(v.x * s) |
                   ((unsigned int)f2bf(v.y * s) << 16);
  dst[(size_t)row * DW + lane] = p;
}

// Issue one TDM transfer: CHUNK codebook rows (32 dwords each) -> LDS with
// 4-dword padding after every 32 dwords (stride 36). data_size = 4B units.
static __device__ inline void tdm_load_chunk(const unsigned int* gsrc,
                                             unsigned ldsOff) {
  const unsigned long long ga = (unsigned long long)(const void*)gsrc;
  v4ui g0;
  g0[0] = 1u;                                        // count=1, user mode
  g0[1] = ldsOff;                                    // LDS byte address
  g0[2] = (unsigned)(ga & 0xFFFFFFFFu);              // global_addr[31:0]
  g0[3] = (unsigned)((ga >> 32) & 0x1FFFFFFu)        // global_addr[56:32]
          | (2u << 30);                              // type = 2 ("image")
  v8i g1;
  g1[0] = (int)((2u << 16)        // data_size = 4 bytes
                | (1u << 20)      // pad_enable
                | (4u << 22)      // pad_interval: every 32 dwords
                | (3u << 25));    // pad_amount: 4 dwords
  g1[1] = (int)(32u << 16);       // tensor_dim0 = 32 (low 16)
  g1[2] = (int)(8192u << 16);     // tensor_dim1 = 8192 (low 16)
  g1[3] = (int)(32u << 16);       // tile_dim0 = 32
  g1[4] = (int)(CHUNK);           // tile_dim1 = 128, tile_dim2 = 0
  g1[5] = 32;                     // tensor_dim0_stride = 32 dwords
  g1[6] = 0;
  g1[7] = 0;
  const v4i gz4 = {0, 0, 0, 0};                      // groups 2/3 unused (2D)
  const v8i gz8 = {0, 0, 0, 0, 0, 0, 0, 0};
  __builtin_amdgcn_tensor_load_to_lds(g0, g1, gz4, gz4, gz8, 0);
}

// 8 waves/block; each wave: 32 tokens vs all 8192 codes via bf16 WMMA.
__global__ __launch_bounds__(256) void vq_argmax(
    const unsigned int* __restrict__ zbf,   // [NTOK][DW] packed bf16
    const unsigned int* __restrict__ cbf,   // [NCODES][DW] packed bf16
    int* __restrict__ idxOut) {
  __shared__ alignas(16) unsigned int ldsB[2 * CH_DWORDS];

  const int tid  = threadIdx.x;
  const int lane = tid & 31;
  const int wave = tid >> 5;
  const int l15  = lane & 15;
  const int hi   = (lane >> 4) & 1;          // half-wave id
  const int j0   = hi * 4;                   // K-dword base per ISA A/B layout
  const int tokBase = (blockIdx.x * 8 + wave) * 32;

  // --- A fragments: two 16-token tiles x 64 (two K=32 halves each).
  const unsigned int* arow0 = zbf + (size_t)(tokBase + l15) * DW;
  const unsigned int* arow1 = arow0 + 16 * DW;
  FragU a0, a1, a2, a3;
  a0.q[0] = *(const v4ui*)(arow0 + j0);
  a0.q[1] = *(const v4ui*)(arow0 + j0 + 8);
  a1.q[0] = *(const v4ui*)(arow0 + 16 + j0);
  a1.q[1] = *(const v4ui*)(arow0 + 16 + j0 + 8);
  a2.q[0] = *(const v4ui*)(arow1 + j0);
  a2.q[1] = *(const v4ui*)(arow1 + j0 + 8);
  a3.q[0] = *(const v4ui*)(arow1 + 16 + j0);
  a3.q[1] = *(const v4ui*)(arow1 + 16 + j0 + 8);

  float best0[8], best1[8];
  int   bidx0[8], bidx1[8];
#pragma unroll
  for (int r = 0; r < 8; ++r) {
    best0[r] = -3.0e38f; bidx0[r] = 0;
    best1[r] = -3.0e38f; bidx1[r] = 0;
  }

  const unsigned ldsBase = (unsigned)(unsigned long long)(const void*)&ldsB[0];

  // Prologue: DMA chunk 0 into buffer 0.
  if (wave == 0) tdm_load_chunk(cbf, ldsBase);

  for (int c = 0; c < NCHUNK; ++c) {
    if (wave == 0) __builtin_amdgcn_s_wait_tensorcnt((short)0);
    __syncthreads();                         // buf[c&1] ready for all waves

    // Overlap: DMA next chunk into the other buffer while we compute.
    if (wave == 0 && (c + 1) < NCHUNK)
      tdm_load_chunk(cbf + (size_t)(c + 1) * CHUNK * DW,
                     ldsBase + ((unsigned)((c + 1) & 1)) * (CH_DWORDS * 4u));

    const unsigned int* buf = &ldsB[(c & 1) * CH_DWORDS];
#pragma unroll 2
    for (int t = 0; t < CHUNK / 16; ++t) {
      // B fragments: column N = l15, same K-dword striping as A; b128 loads.
      const unsigned int* brow = buf + (t * 16 + l15) * LDS_STRIDE;
      FragU b0, b1;
      b0.q[0] = *(const v4ui*)(brow + j0);
      b0.q[1] = *(const v4ui*)(brow + j0 + 8);
      b1.q[0] = *(const v4ui*)(brow + 16 + j0);
      b1.q[1] = *(const v4ui*)(brow + 16 + j0 + 8);

      v8f c0 = {}, c1 = {};
      c0 = __builtin_amdgcn_wmma_f32_16x16x32_bf16(false, a0.h, false, b0.h,
                                                   (short)0, c0, false, false);
      c0 = __builtin_amdgcn_wmma_f32_16x16x32_bf16(false, a1.h, false, b1.h,
                                                   (short)0, c0, false, false);
      c1 = __builtin_amdgcn_wmma_f32_16x16x32_bf16(false, a2.h, false, b0.h,
                                                   (short)0, c1, false, false);
      c1 = __builtin_amdgcn_wmma_f32_16x16x32_bf16(false, a3.h, false, b1.h,
                                                   (short)0, c1, false, false);

      const int n = c * CHUNK + t * 16 + l15;
#pragma unroll
      for (int r = 0; r < 8; ++r) {
        const float v0 = c0[r];
        const bool g0 = (v0 > best0[r]);      // strict: keeps earliest index
        best0[r] = g0 ? v0 : best0[r];
        bidx0[r] = g0 ? n  : bidx0[r];
        const float v1 = c1[r];
        const bool g1 = (v1 > best1[r]);
        best1[r] = g1 ? v1 : best1[r];
        bidx1[r] = g1 ? n  : bidx1[r];
      }
    }
    __syncthreads();                         // all waves done with buf[c&1]
  }

  // Cross-lane argmax within each 16-lane half (C layout: VGPR r = row r/r+8).
#pragma unroll
  for (int r = 0; r < 8; ++r) {
#pragma unroll
    for (int m = 8; m >= 1; m >>= 1) {
      float ov = __shfl_xor(best0[r], m, 32);
      int   oi = __shfl_xor(bidx0[r], m, 32);
      if (ov > best0[r] || (ov == best0[r] && oi < bidx0[r])) {
        best0[r] = ov; bidx0[r] = oi;
      }
      ov = __shfl_xor(best1[r], m, 32);
      oi = __shfl_xor(bidx1[r], m, 32);
      if (ov > best1[r] || (ov == best1[r] && oi < bidx1[r])) {
        best1[r] = ov; bidx1[r] = oi;
      }
    }
  }
  if (l15 == 0) {
#pragma unroll
    for (int r = 0; r < 8; ++r) {
      idxOut[tokBase + hi * 8 + r]      = bidx0[r];
      idxOut[tokBase + 16 + hi * 8 + r] = bidx1[r];
    }
  }
}

// One wave per token: write z_qnorm, accumulate loss, emit idx as float.
__global__ __launch_bounds__(256) void vq_finalize(
    const float* __restrict__ z, const float* __restrict__ cb,
    const int* __restrict__ idx, float* __restrict__ out) {
  const int wave = threadIdx.x >> 5;
  const int lane = threadIdx.x & 31;
  const int t = blockIdx.x * 8 + wave;

  const float2 zv = ((const float2*)(z + (size_t)t * DVEC))[lane];
  float ssz = zv.x * zv.x + zv.y * zv.y;
  const int n = idx[t];
  const float2 qv = ((const float2*)(cb + (size_t)n * DVEC))[lane];
  float ssq = qv.x * qv.x + qv.y * qv.y;
#pragma unroll
  for (int m = 16; m >= 1; m >>= 1) {
    ssz += __shfl_xor(ssz, m, 32);
    ssq += __shfl_xor(ssq, m, 32);
  }
  const float sz = 1.0f / fmaxf(sqrtf(ssz), 1e-12f);
  const float sq = 1.0f / fmaxf(sqrtf(ssq), 1e-12f);

  float2 zq; zq.x = qv.x * sq; zq.y = qv.y * sq;
  ((float2*)(out + (size_t)t * DVEC))[lane] = zq;   // z_q_out == z_qnorm (fwd)

  const float dx = zq.x - zv.x * sz;
  const float dy = zq.y - zv.y * sz;
  float ls = dx * dx + dy * dy;
#pragma unroll
  for (int m = 16; m >= 1; m >>= 1) ls += __shfl_xor(ls, m, 32);
  if (lane == 0) {
    atomicAdd(out + (size_t)NTOK * DVEC,
              ls * (1.25f / (float)((size_t)NTOK * DVEC)));
    out[(size_t)NTOK * DVEC + 1 + t] = (float)n;    // idx, as f32 payload
  }
}

extern "C" void kernel_launch(void* const* d_in, const int* in_sizes, int n_in,
                              void* d_out, int out_size, void* d_ws,
                              size_t ws_size, hipStream_t stream) {
  const float* z  = (const float*)d_in[0];   // [8,64,64,64]
  const float* cb = (const float*)d_in[1];   // [8192,64]
  float* out = (float*)d_out;                // [zq(2097152) | loss(1) | idx(32768)]
  char* ws = (char*)d_ws;

  unsigned int* cbf = (unsigned int*)ws;                               // 1 MB
  unsigned int* zbf = (unsigned int*)(ws + (size_t)NCODES * DW * 4);   // 4 MB
  int* idx = (int*)(ws + (size_t)NCODES * DW * 4 + (size_t)NTOK * DW * 4);
  float* lossPtr = out + (size_t)NTOK * DVEC;

  nrm_rows<<<NCODES / 8, 256, 0, stream>>>(cb, cbf, NCODES, lossPtr);
  nrm_rows<<<NTOK / 8, 256, 0, stream>>>(z, zbf, NTOK, nullptr);
  vq_argmax<<<NTOK / 256, 256, 0, stream>>>(zbf, cbf, idx);
  vq_finalize<<<NTOK / 8, 256, 0, stream>>>(z, cb, idx, out);
}
